// MRTransformerPredictor_68771016344163
// MI455X (gfx1250) — matrix operand order repaired and checked by reading, not compiled, not run on hardware
//
#include <hip/hip_runtime.h>
#include <hip/hip_bf16.h>
#include <math.h>

// ---------------------------------------------------------------------------
// MRTransformerPredictor forward on MI455X (gfx1250, wave32, WMMA).
// All matmuls run through v_wmma_f32_16x16x32_f16 (f32 in/out, f16 operands,
// f32 accumulate). Encoder/decoder are batch-invariant (computed once); only
// the last conv position and the 5 shifted column means of x are needed.
// GEMM: block = 4 waves = 64x64 C macro-tile. B k-slab staged once per block
// in LDS (f16, fragment-ready layout); each wave: 1 A fragment (global),
// 4 B fragments loaded up-front (8x ds_load_b128, one dscnt wait), then
// 4 back-to-back WMMAs per 32-wide K step (XDL co-execution friendly).
// Requires K % 32 == 0 (true for every call site: 64/256/512/1024/2048).
// ---------------------------------------------------------------------------

typedef __attribute__((ext_vector_type(16))) _Float16 v16h;
typedef __attribute__((ext_vector_type(8)))  float    v8f;

#define Dm   512
#define Sm   256
#define Hm   8
#define DHm  64
#define DFFm 2048
#define NFm  64
#define Bm   16

// Bs[n][p] holds the packed f16 pair (k=2p, k=2p+1) of column nb0+n.
// Row stride 20 dwords: 80B (16B aligned) and 20n mod 64 distinct for n=0..15.
#define BS_STRIDE 20

template <int TB>
__global__ void k_gemm(const float* __restrict__ A, int lda,
                       const float* __restrict__ B, int ldb,
                       const float* __restrict__ bias,
                       float* __restrict__ C, int ldc,
                       int M, int N, int K, int act)
{
    __shared__ unsigned int Bs[64][BS_STRIDE];

    const int tid  = threadIdx.x;
    const int lane = tid & 31;
    const int wave = tid >> 5;
    const int tm   = blockIdx.y * 4 + wave;   // wave-uniform M tile
    const int nb0  = blockIdx.x * 64;         // block's 64-wide N strip

    const int  hl  = lane >> 4;               // half-lane 0/1
    const int  l15 = lane & 15;
    const bool rowActive = (tm * 16 < M);     // wave-uniform
    const int  ar  = min(tm * 16 + l15, M - 1);

    v8f acc[4];
#pragma unroll
    for (int t = 0; t < 4; ++t) acc[t] = (v8f){0.f,0.f,0.f,0.f,0.f,0.f,0.f,0.f};

    for (int kb = 0; kb < K; kb += 32) {
        __syncthreads();
        // ---- cooperative B k-slab stage: 32(K) x 64(N) f32 -> f16 in LDS ---
        if (TB) {
            // B stored [N,K]: pair (n,p) at B[(nb0+n)*ldb + kb + 2p]
#pragma unroll
            for (int r = 0; r < 8; ++r) {
                int q = tid + r * 128;                 // pair index 0..1023
                int n = q >> 4, p = q & 15;
                int gn = min(nb0 + n, N - 1);
                const float2 bv = *(const float2*)(B + (size_t)gn * ldb + kb + 2 * p);
                union { unsigned int u; _Float16 h[2]; } pk;
                pk.h[0] = (_Float16)bv.x;
                pk.h[1] = (_Float16)bv.y;
                Bs[n][p] = pk.u;
                if (r == 0 && kb + 32 < K)
                    __builtin_prefetch(B + (size_t)gn * ldb + kb + 32, 0, 1);
            }
        } else {
            // B stored [K,N]: element (k,n) at B[(kb+k)*ldb + nb0+n]
#pragma unroll
            for (int r = 0; r < 16; ++r) {
                int q = tid + r * 128;                 // element 0..2047
                int k = q >> 6, n = q & 63;
                int gn = min(nb0 + n, N - 1);
                float v = B[(size_t)(kb + k) * ldb + gn];
                ((_Float16*)&Bs[n][0])[k] = (_Float16)v;
                if (r == 0 && kb + 32 < K)
                    __builtin_prefetch(B + (size_t)(kb + 32) * ldb + gn, 0, 1);
            }
        }
        __syncthreads();

        if (rowActive) {
            // ---- A fragment: VGPR j holds K pair (j>>2)*16 + hl*8 + (j&3)*2
            v16h a;
#pragma unroll
            for (int j = 0; j < 8; ++j) {
                int ka = kb + ((j >> 2) << 4) + (hl << 3) + ((j & 3) << 1);
                const float2 av = *(const float2*)(A + (size_t)ar * lda + ka);
                a[2 * j]     = (_Float16)av.x;
                a[2 * j + 1] = (_Float16)av.y;
            }
            // ---- load all 4 B fragments first (one dscnt wait), ...
            union { v16h v; uint4 q[2]; } bu[4];
#pragma unroll
            for (int t = 0; t < 4; ++t) {
                const unsigned int* row = &Bs[t * 16 + l15][0];
                bu[t].q[0] = *(const uint4*)(row + hl * 4);
                bu[t].q[1] = *(const uint4*)(row + 8 + hl * 4);
            }
            // ---- ... then 4 back-to-back WMMAs (independent accumulators)
            acc[0] = __builtin_amdgcn_wmma_f32_16x16x32_f16(
                         false, a, false, bu[0].v, (short)0, acc[0], false, false);
            acc[1] = __builtin_amdgcn_wmma_f32_16x16x32_f16(
                         false, a, false, bu[1].v, (short)0, acc[1], false, false);
            acc[2] = __builtin_amdgcn_wmma_f32_16x16x32_f16(
                         false, a, false, bu[2].v, (short)0, acc[2], false, false);
            acc[3] = __builtin_amdgcn_wmma_f32_16x16x32_f16(
                         false, a, false, bu[3].v, (short)0, acc[3], false, false);
        }
    }

    if (rowActive) {
#pragma unroll
        for (int t = 0; t < 4; ++t) {
#pragma unroll
            for (int i = 0; i < 8; ++i) {
                int m = tm * 16 + (hl << 3) + i;   // C/D: VGPR i, halflane -> +8
                int n = nb0 + t * 16 + l15;
                if (m < M && n < N) {
                    float v = acc[t][i];
                    if (bias) v += bias[n];
                    if (act == 1) v = v > 0.f ? v : 0.f;   // relu
                    else if (act == 2) v = tanhf(v);       // tanh
                    C[(size_t)m * ldc + n] = v;
                }
            }
        }
    }
}

// ---------------------------------------------------------------------------
// Encoder input: mean(node_emb) ++ mean(param_emb) + sinusoidal PE. [S,D]
// ---------------------------------------------------------------------------
__global__ void k_enc_init(float* __restrict__ enc,
                           const float* __restrict__ node_emb,
                           const float* __restrict__ param_emb)
{
    int idx = blockIdx.x * blockDim.x + threadIdx.x;
    if (idx >= Sm * Dm) return;
    int s = idx >> 9, d = idx & (Dm - 1);
    float e;
    if (d < 256) {
        float a = 0.f;
        for (int n = 0; n < 16; ++n) a += node_emb[n * 256 + d];
        e = a * (1.f / 16.f);
    } else {
        float a = 0.f;
        for (int p = 0; p < 4; ++p) a += param_emb[p * 256 + (d - 256)];
        e = a * 0.25f;
    }
    int j = d >> 1;
    float div = __expf((float)(2 * j) * (-logf(10000.f) / (float)Dm));
    float ang = (float)s * div;
    float pe = (d & 1) ? __cosf(ang) : __sinf(ang);
    enc[idx] = e + pe;
}

// ---------------------------------------------------------------------------
// Row softmax over N with optional same-segment mask and 1/sqrt(dh) scale.
// Rows laid out [H, Sq, N]; q = row % Sq is used for the seg mask.
// ---------------------------------------------------------------------------
__global__ void k_softmax(float* __restrict__ s, int N, int Sq,
                          const int* __restrict__ seg, float scale)
{
    int r = blockIdx.x;
    int q = r % Sq;
    float* row = s + (size_t)r * N;
    __shared__ float red[256];
    int tid = threadIdx.x;
    int sq = seg ? seg[q] : 0;

    float mx = -1e30f;
    for (int k = tid; k < N; k += blockDim.x) {
        float v = row[k] * scale;
        if (seg && seg[k] != sq) v = -1e9f;
        row[k] = v;
        mx = fmaxf(mx, v);
    }
    red[tid] = mx; __syncthreads();
    for (int o = blockDim.x >> 1; o > 0; o >>= 1) {
        if (tid < o) red[tid] = fmaxf(red[tid], red[tid + o]);
        __syncthreads();
    }
    mx = red[0]; __syncthreads();

    float sum = 0.f;
    for (int k = tid; k < N; k += blockDim.x) {
        float e = __expf(row[k] - mx);
        row[k] = e;
        sum += e;
    }
    red[tid] = sum; __syncthreads();
    for (int o = blockDim.x >> 1; o > 0; o >>= 1) {
        if (tid < o) red[tid] += red[tid + o];
        __syncthreads();
    }
    float inv = 1.f / red[0];
    for (int k = tid; k < N; k += blockDim.x) row[k] *= inv;
}

// x = LayerNorm(x + y) * g + b   (in-place on x), one block per row.
__global__ void k_add_ln(float* __restrict__ x, const float* __restrict__ y,
                         const float* __restrict__ g, const float* __restrict__ b,
                         int D)
{
    int r = blockIdx.x;
    __shared__ float red[256];
    int tid = threadIdx.x;
    float s1 = 0.f, s2 = 0.f;
    for (int d = tid; d < D; d += blockDim.x) {
        float v = x[(size_t)r * D + d] + y[(size_t)r * D + d];
        s1 += v; s2 += v * v;
    }
    red[tid] = s1; __syncthreads();
    for (int o = blockDim.x >> 1; o > 0; o >>= 1) { if (tid < o) red[tid] += red[tid + o]; __syncthreads(); }
    float mean = red[0] / (float)D; __syncthreads();
    red[tid] = s2; __syncthreads();
    for (int o = blockDim.x >> 1; o > 0; o >>= 1) { if (tid < o) red[tid] += red[tid + o]; __syncthreads(); }
    float var = red[0] / (float)D - mean * mean; __syncthreads();
    float inv = rsqrtf(var + 1e-5f);
    for (int d = tid; d < D; d += blockDim.x) {
        float v = x[(size_t)r * D + d] + y[(size_t)r * D + d];
        x[(size_t)r * D + d] = (v - mean) * inv * g[d] + b[d];
    }
}

// seg[q] = argmax_p  x[q,:] . proto[p,:]   (first max wins, like jnp.argmax)
__global__ void k_seg(const float* __restrict__ x, const float* __restrict__ proto,
                      int S, int* __restrict__ seg)
{
    int q = blockIdx.x * blockDim.x + threadIdx.x;
    if (q >= S) return;
    float best = -1e30f; int bi = 0;
    for (int p = 0; p < 8; ++p) {
        float acc = 0.f;
        for (int d = 0; d < Dm; ++d) acc += x[(size_t)q * Dm + d] * proto[p * Dm + d];
        if (acc > best) { best = acc; bi = p; }
    }
    seg[q] = bi;
}

__global__ void k_counts(const int* __restrict__ seg, int* __restrict__ cnt, int S)
{
    int q = blockIdx.x * blockDim.x + threadIdx.x;
    if (q >= S) return;
    int c = 0, sq = seg[q];
    for (int k = 0; k < S; ++k) c += (seg[k] == sq);
    cnt[q] = c;
}

// dst[r*ldd + doff + d] = src[r*D + d]  where counts[r]==1 (singleton segments)
__global__ void k_override(float* __restrict__ dst, int ldd, int doff,
                           const float* __restrict__ src,
                           const int* __restrict__ cnt, int R, int D)
{
    int idx = blockIdx.x * blockDim.x + threadIdx.x;
    if (idx >= R * D) return;
    int r = idx / D, d = idx - r * D;
    if (cnt[r] == 1) dst[(size_t)r * ldd + doff + d] = src[(size_t)r * D + d];
}

__global__ void k_copy(float* __restrict__ dst, const float* __restrict__ src, int n)
{
    int i = blockIdx.x * blockDim.x + threadIdx.x;
    if (i < n) dst[i] = src[i];
}

// m5[bf,k] = mean_s x_pad[b, s+k-2, f]  via total column sum minus edges.
__global__ void k_m5(const float* __restrict__ x, float* __restrict__ m5)
{
    int bf = blockIdx.x * blockDim.x + threadIdx.x;
    if (bf >= Bm * NFm) return;
    int b = bf >> 6, f = bf & 63;
    const float* xb = x + (size_t)b * Sm * NFm + f;
    float tot = 0.f;
    for (int s = 0; s < Sm; ++s) tot += xb[(size_t)s * NFm];
    float x0 = xb[0], x1 = xb[NFm];
    float xm2 = xb[(size_t)(Sm - 2) * NFm], xm1 = xb[(size_t)(Sm - 1) * NFm];
    const float inv = 1.f / (float)Sm;
    float* m = m5 + (size_t)bf * 5;
    m[0] = (tot - xm2 - xm1) * inv;
    m[1] = (tot - xm1) * inv;
    m[2] = tot * inv;
    m[3] = (tot - x0) * inv;
    m[4] = (tot - x0 - x1) * inv;
}

// mfeat[bf,d] = sum_k cw[(f*D+d)*5+k]*m5[bf,k] + conv_b[f*D+d]
__global__ void k_mfeat(const float* __restrict__ m5, const float* __restrict__ cw,
                        const float* __restrict__ cb, float* __restrict__ mf)
{
    int idx = blockIdx.x * blockDim.x + threadIdx.x;
    if (idx >= Bm * NFm * Dm) return;
    int d = idx & (Dm - 1);
    int bf = idx >> 9;
    int f = bf & 63;
    const float* w = cw + (size_t)(f * Dm + d) * 5;
    const float* m = m5 + (size_t)bf * 5;
    mf[idx] = w[0]*m[0] + w[1]*m[1] + w[2]*m[2] + w[3]*m[3] + w[4]*m[4]
            + cb[f * Dm + d];
}

// score[r] = t[r,:] . aw2 + ab2
__global__ void k_rowdot(const float* __restrict__ t, const float* __restrict__ aw2,
                         const float* __restrict__ ab2, float* __restrict__ out, int R)
{
    int r = blockIdx.x * blockDim.x + threadIdx.x;
    if (r >= R) return;
    float acc = 0.f;
    for (int d = 0; d < Dm; ++d) acc += t[(size_t)r * Dm + d] * aw2[d];
    out[r] = acc + ab2[0];
}

// softmax over F=64 per batch row
__global__ void k_softmax_f(const float* __restrict__ sc, float* __restrict__ w)
{
    int b = blockIdx.x * blockDim.x + threadIdx.x;
    if (b >= Bm) return;
    float mx = -1e30f;
    for (int f = 0; f < NFm; ++f) mx = fmaxf(mx, sc[b * NFm + f]);
    float sum = 0.f;
    for (int f = 0; f < NFm; ++f) { float e = __expf(sc[b * NFm + f] - mx); w[b * NFm + f] = e; sum += e; }
    float inv = 1.f / sum;
    for (int f = 0; f < NFm; ++f) w[b * NFm + f] *= inv;
}

// vf_raw[b,d] = sum_f w[b,f]*( sum_{k=0..2} cw[(f*D+d)*5+k]*x[b,S-3+k,f] + cb[f,d] )
__global__ void k_vfraw(const float* __restrict__ x, const float* __restrict__ cw,
                        const float* __restrict__ cb, const float* __restrict__ w,
                        float* __restrict__ vf)
{
    int idx = blockIdx.x * blockDim.x + threadIdx.x;
    if (idx >= Bm * Dm) return;
    int b = idx >> 9, d = idx & (Dm - 1);
    float acc = 0.f;
    for (int f = 0; f < NFm; ++f) {
        const float* cwp = cw + (size_t)(f * Dm + d) * 5;
        const float* xb = x + (size_t)b * Sm * NFm + f;
        float conv = cb[f * Dm + d]
                   + cwp[0] * xb[(size_t)(Sm - 3) * NFm]
                   + cwp[1] * xb[(size_t)(Sm - 2) * NFm]
                   + cwp[2] * xb[(size_t)(Sm - 1) * NFm];
        acc += w[b * NFm + f] * conv;
    }
    vf[idx] = acc;
}

// comb[b, 0:512] = dec0 ; comb[b, 512:1024] = vf[b]
__global__ void k_comb(const float* __restrict__ dec, const float* __restrict__ vf,
                       float* __restrict__ comb)
{
    int idx = blockIdx.x * blockDim.x + threadIdx.x;
    if (idx >= Bm * 2 * Dm) return;
    int b = idx >> 10, j = idx & 1023;
    comb[idx] = (j < Dm) ? dec[j] : vf[(size_t)b * Dm + (j - Dm)];
}

// ---------------------------------------------------------------------------
// Host orchestration
// ---------------------------------------------------------------------------
extern "C" void kernel_launch(void* const* d_in, const int* in_sizes, int n_in,
                              void* d_out, int out_size, void* d_ws, size_t ws_size,
                              hipStream_t stream)
{
    (void)in_sizes; (void)n_in; (void)out_size; (void)ws_size;
    auto P = [&](int i) { return (const float*)d_in[i]; };

    // jax-style flatten (sorted dict keys, DFS, lists in order):
    // 0 ab1, 1 ab2, 2 aw1, 3 aw2, 4 conv_b, 5 conv_w,
    // 6 cross.b_in, 7 cross.b_out, 8 cross.w_in, 9 cross.w_out,
    // dec layer L base 10+21L, enc layer L base 73+19L (see below),
    // 130 int_b, 131 int_w, 132 node_emb, 133 out_b, 134 out_w, 135 param_emb,
    // 136 query, 137 vproj_b, 138 vproj_w, 139 x
    const float* ab1     = P(0);
    const float* ab2     = P(1);
    const float* aw1     = P(2);
    const float* aw2     = P(3);
    const float* conv_b  = P(4);
    const float* conv_w  = P(5);
    const float* cr_bin  = P(6);
    const float* cr_bout = P(7);
    const float* cr_win  = P(8);
    const float* cr_wout = P(9);
    const float* int_b   = P(130);
    const float* int_w   = P(131);
    const float* node_e  = P(132);
    const float* out_b   = P(133);
    const float* out_w   = P(134);
    const float* param_e = P(135);
    const float* query   = P(136);
    const float* vproj_b = P(137);
    const float* vproj_w = P(138);
    const float* x       = P(139);

    // --- workspace carve ---
    size_t off = 0;
    auto alloc = [&](size_t bytes) {
        void* p = (char*)d_ws + off;
        off += (bytes + 255) & ~(size_t)255;
        return p;
    };
    float* ENC  = (float*)alloc(Sm * Dm * 4);
    float* Qb   = (float*)alloc(Sm * Dm * 4);
    float* Kb   = (float*)alloc(Sm * Dm * 4);
    float* Vb   = (float*)alloc(Sm * Dm * 4);
    float* SC   = (float*)alloc((size_t)Hm * Sm * Sm * 4);
    float* Ob   = (float*)alloc(Sm * Dm * 4);
    float* CAT  = (float*)alloc(Sm * 2 * Dm * 4);
    float* TMP  = (float*)alloc(Sm * Dm * 4);
    float* FFH  = (float*)alloc(Sm * DFFm * 4);
    int*   SEG  = (int*)alloc(Sm * 4);
    int*   CNT  = (int*)alloc(Sm * 4);
    float* KC   = (float*)alloc(Sm * Dm * 4);
    float* VC   = (float*)alloc(Sm * Dm * 4);
    float* DEC  = (float*)alloc(Dm * 4);
    float* DECV = (float*)alloc(Dm * 4);
    float* DCAT = (float*)alloc(2 * Dm * 4);
    float* DTMP = (float*)alloc(Dm * 4);
    float* DQ   = (float*)alloc(Dm * 4);
    float* DSC  = (float*)alloc(Hm * Sm * 4);
    float* DO   = (float*)alloc(Dm * 4);
    float* DFH  = (float*)alloc(DFFm * 4);
    float* MF   = (float*)alloc((size_t)Bm * NFm * Dm * 4);
    float* TT   = (float*)alloc((size_t)Bm * NFm * Dm * 4);
    float* M5   = (float*)alloc(Bm * NFm * 5 * 4);
    float* SCR  = (float*)alloc(Bm * NFm * 4);
    float* WS   = (float*)alloc(Bm * NFm * 4);
    float* VFR  = (float*)alloc(Bm * Dm * 4);
    float* VF   = (float*)alloc(Bm * Dm * 4);
    float* COMB = (float*)alloc(Bm * 2 * Dm * 4);
    float* INTG = (float*)alloc(Bm * Dm * 4);

    auto gemm = [&](const float* A, int lda, const float* B, int ldb, int tb,
                    const float* bias, float* C, int ldc,
                    int M, int N, int K, int act) {
        dim3 g((N + 63) / 64, (M + 63) / 64);
        if (tb)
            k_gemm<1><<<g, dim3(128), 0, stream>>>(A, lda, B, ldb, bias, C, ldc, M, N, K, act);
        else
            k_gemm<0><<<g, dim3(128), 0, stream>>>(A, lda, B, ldb, bias, C, ldc, M, N, K, act);
    };

    // full self-MHA over S=256; writes out-projection into cdst (ldc, col off)
    auto mha_self = [&](const float* xin, const float* w_in, const float* b_in,
                        const float* w_out, const float* b_out,
                        const int* seg, float* cdst, int cld, int coff) {
        gemm(xin, Dm, w_in,               Dm, 1, b_in,           Qb, Dm, Sm, Dm, Dm, 0);
        gemm(xin, Dm, w_in + Dm * Dm,     Dm, 1, b_in + Dm,      Kb, Dm, Sm, Dm, Dm, 0);
        gemm(xin, Dm, w_in + 2 * Dm * Dm, Dm, 1, b_in + 2 * Dm,  Vb, Dm, Sm, Dm, Dm, 0);
        for (int h = 0; h < Hm; ++h)
            gemm(Qb + h * DHm, Dm, Kb + h * DHm, Dm, 1, nullptr,
                 SC + (size_t)h * Sm * Sm, Sm, Sm, Sm, DHm, 0);
        k_softmax<<<Hm * Sm, 256, 0, stream>>>(SC, Sm, Sm, seg, 0.125f);
        for (int h = 0; h < Hm; ++h)
            gemm(SC + (size_t)h * Sm * Sm, Sm, Vb + h * DHm, Dm, 0, nullptr,
                 Ob + h * DHm, Dm, Sm, DHm, Sm, 0);
        gemm(Ob, Dm, w_out, Dm, 1, b_out, cdst + coff, cld, Sm, Dm, Dm, 0);
    };

    // ---------------- encoder (batch-invariant: computed once) -------------
    k_enc_init<<<(Sm * Dm + 255) / 256, 256, 0, stream>>>(ENC, node_e, param_e);

    for (int L = 0; L < 3; ++L) {
        int EB = 73 + 19 * L;
        const float* ffn_b1 = P(EB + 0), *ffn_b2 = P(EB + 1);
        const float* ffn_w1 = P(EB + 2), *ffn_w2 = P(EB + 3);
        const float* ln1_b = P(EB + 4), *ln1_g = P(EB + 5);
        const float* ln2_b = P(EB + 6), *ln2_g = P(EB + 7);
        const float* fus_b = P(EB + 8), *fus_w = P(EB + 9);
        const float* lbin = P(EB + 10), *lbout = P(EB + 11);
        const float* lwin = P(EB + 12), *lwout = P(EB + 13);
        const float* proto = P(EB + 14);
        const float* sbin = P(EB + 15), *sbout = P(EB + 16);
        const float* swin = P(EB + 17), *swout = P(EB + 18);

        k_seg<<<1, 256, 0, stream>>>(ENC, proto, Sm, SEG);
        k_counts<<<1, 256, 0, stream>>>(SEG, CNT, Sm);
        // long -> CAT[:, :512], short (masked) -> CAT[:, 512:]
        mha_self(ENC, lwin, lbin, lwout, lbout, nullptr, CAT, 2 * Dm, 0);
        mha_self(ENC, swin, sbin, swout, sbout, SEG,     CAT, 2 * Dm, Dm);
        // singleton segments pass raw x through the short branch
        k_override<<<(Sm * Dm + 255) / 256, 256, 0, stream>>>(CAT, 2 * Dm, Dm, ENC, CNT, Sm, Dm);
        // fusion
        gemm(CAT, 2 * Dm, fus_w, 2 * Dm, 1, fus_b, TMP, Dm, Sm, Dm, 2 * Dm, 0);
        k_add_ln<<<Sm, 256, 0, stream>>>(ENC, TMP, ln1_g, ln1_b, Dm);
        // ffn
        gemm(ENC, Dm, ffn_w1, Dm, 1, ffn_b1, FFH, DFFm, Sm, DFFm, Dm, 1);
        gemm(FFH, DFFm, ffn_w2, DFFm, 1, ffn_b2, TMP, Dm, Sm, Dm, DFFm, 0);
        k_add_ln<<<Sm, 256, 0, stream>>>(ENC, TMP, ln2_g, ln2_b, Dm);
    }

    // cross-attention K/V of encoder output (shared by all decoder layers)
    gemm(ENC, Dm, cr_win + Dm * Dm,     Dm, 1, cr_bin + Dm,     KC, Dm, Sm, Dm, Dm, 0);
    gemm(ENC, Dm, cr_win + 2 * Dm * Dm, Dm, 1, cr_bin + 2 * Dm, VC, Dm, Sm, Dm, Dm, 0);

    // ---------------- decoder (batch-invariant, seq len 1) -----------------
    k_copy<<<2, 256, 0, stream>>>(DEC, query, Dm);
    for (int L = 0; L < 3; ++L) {
        int DB = 10 + 21 * L;
        const float* ffn_b1 = P(DB + 0), *ffn_b2 = P(DB + 1);
        const float* ffn_w1 = P(DB + 2), *ffn_w2 = P(DB + 3);
        const float* ln1_b = P(DB + 4), *ln1_g = P(DB + 5);
        const float* ln2_b = P(DB + 6), *ln2_g = P(DB + 7);
        const float* ln3_b = P(DB + 8), *ln3_g = P(DB + 9);
        const float* fus_b = P(DB + 10), *fus_w = P(DB + 11);
        const float* lbin = P(DB + 12), *lbout = P(DB + 13);
        const float* lwin = P(DB + 14), *lwout = P(DB + 15);
        // proto/short params unused: seq len 1 => counts==1 => short branch = x

        // long branch, seq len 1: softmax over one key == 1 => o = V(dec)
        gemm(DEC, Dm, lwin + 2 * Dm * Dm, Dm, 1, lbin + 2 * Dm, DECV, Dm, 1, Dm, Dm, 0);
        gemm(DECV, Dm, lwout, Dm, 1, lbout, DCAT, 2 * Dm, 1, Dm, Dm, 0);
        // short branch: singleton segment -> raw dec
        k_copy<<<2, 256, 0, stream>>>(DCAT + Dm, DEC, Dm);
        gemm(DCAT, 2 * Dm, fus_w, 2 * Dm, 1, fus_b, DTMP, Dm, 1, Dm, 2 * Dm, 0);
        k_add_ln<<<1, 256, 0, stream>>>(DEC, DTMP, ln1_g, ln1_b, Dm);

        // cross attention: 1 query over S=256 encoder keys
        gemm(DEC, Dm, cr_win, Dm, 1, cr_bin, DQ, Dm, 1, Dm, Dm, 0);
        for (int h = 0; h < Hm; ++h)
            gemm(DQ + h * DHm, Dm, KC + h * DHm, Dm, 1, nullptr,
                 DSC + h * Sm, Sm, 1, Sm, DHm, 0);
        k_softmax<<<Hm, 256, 0, stream>>>(DSC, Sm, 1, nullptr, 0.125f);
        for (int h = 0; h < Hm; ++h)
            gemm(DSC + h * Sm, Sm, VC + h * DHm, Dm, 0, nullptr,
                 DO + h * DHm, Dm, 1, DHm, Sm, 0);
        gemm(DO, Dm, cr_wout, Dm, 1, cr_bout, DTMP, Dm, 1, Dm, Dm, 0);
        k_add_ln<<<1, 256, 0, stream>>>(DEC, DTMP, ln2_g, ln2_b, Dm);

        // ffn
        gemm(DEC, Dm, ffn_w1, Dm, 1, ffn_b1, DFH, DFFm, 1, DFFm, Dm, 1);
        gemm(DFH, DFFm, ffn_w2, DFFm, 1, ffn_b2, DTMP, Dm, 1, Dm, DFFm, 0);
        k_add_ln<<<1, 256, 0, stream>>>(DEC, DTMP, ln3_g, ln3_b, Dm);
    }

    // ---------------- variable-specific conv path (batch-dependent) -------
    k_m5<<<(Bm * NFm + 255) / 256, 256, 0, stream>>>(x, M5);
    k_mfeat<<<(Bm * NFm * Dm + 255) / 256, 256, 0, stream>>>(M5, conv_w, conv_b, MF);
    gemm(MF, Dm, aw1, Dm, 1, ab1, TT, Dm, Bm * NFm, Dm, Dm, 2 /*tanh*/);
    k_rowdot<<<(Bm * NFm + 255) / 256, 256, 0, stream>>>(TT, aw2, ab2, SCR, Bm * NFm);
    k_softmax_f<<<1, 64, 0, stream>>>(SCR, WS);
    k_vfraw<<<(Bm * Dm + 255) / 256, 256, 0, stream>>>(x, conv_w, conv_b, WS, VFR);
    gemm(VFR, Dm, vproj_w, Dm, 1, vproj_b, VF, Dm, Bm, Dm, Dm, 0);

    // ---------------- head --------------------------------------------------
    k_comb<<<(Bm * 2 * Dm + 255) / 256, 256, 0, stream>>>(DEC, VF, COMB);
    gemm(COMB, 2 * Dm, int_w, 2 * Dm, 1, int_b, INTG, Dm, Bm, Dm, 2 * Dm, 0);
    gemm(INTG, Dm, out_w, Dm, 1, out_b, (float*)d_out, NFm, Bm, NFm, Dm, 0);
}
